// TransformerBlock_28578712388009
// MI455X (gfx1250) — matrix operand order repaired and checked by reading, not compiled, not run on hardware
//
#include <hip/hip_runtime.h>

#define SEQ   25
#define HEADS 8
#define DCH   128

typedef __attribute__((ext_vector_type(16))) __bf16 v16bf;
typedef __attribute__((ext_vector_type(8)))  float  v8f;

__device__ inline unsigned short f2bf(float f) {
  unsigned int u = __float_as_uint(f);
  u += 0x7FFFu + ((u >> 16) & 1u);          // round-to-nearest-even
  return (unsigned short)(u >> 16);
}

// ---------------------------------------------------------------------------
// Tensor Data Mover staging (gfx1250). Loads a 1 x 32 x 16 tile of 4-byte
// elements (dim0 = 1 contiguous elem, dim1 = 32 strided "k", dim2 = 16 rows)
// into LDS with +1-DWORD padding after every 32 DWORDs (row stride = 33).
// ---------------------------------------------------------------------------
#if defined(__has_builtin)
#if __has_builtin(__builtin_amdgcn_tensor_load_to_lds) && \
    __has_builtin(__builtin_amdgcn_s_wait_tensorcnt)
#define HAVE_TDM 1
#endif
#endif

#ifdef HAVE_TDM
typedef unsigned int u32x4 __attribute__((ext_vector_type(4)));
typedef int          i32x4 __attribute__((ext_vector_type(4)));
typedef int          i32x8 __attribute__((ext_vector_type(8)));

__device__ inline void tdm_load_tile(const float* gsrc, float* lds_dst,
                                     unsigned long long stride0_elems,
                                     unsigned long long stride1_elems)
{
  unsigned long long ga = (unsigned long long)gsrc;     // 57-bit global byte addr
  unsigned la = (unsigned)(unsigned long long)lds_dst;  // LDS byte addr (low 32)
  const unsigned TD = 0x7FFFFFFFu;                      // huge dims: no OOB clamp

  // D# group 0: count=1 | lds_addr | global_addr[56:0] | type=2 ("image")
  u32x4 g0 = { 1u, la, (unsigned)ga,
               (unsigned)((ga >> 32) & 0x01FFFFFFull) | (2u << 30) };

  // D# group 1
  i32x8 g1;
  g1[0] = (int)((2u << 16)            // data_size = 4 bytes
              | (1u << 20)            // pad_enable
              | (4u << 22));          // pad_interval code 4 -> every 32 DWORDs
                                      // pad_amount  code 0 -> +1 DWORD
  g1[1] = (int)((TD & 0xFFFFu) << 16);                         // tensor_dim0[15:0]
  g1[2] = (int)(((TD >> 16) & 0xFFFFu) | ((TD & 0xFFFFu) << 16));   // td0[31:16]|td1[15:0]
  g1[3] = (int)(((TD >> 16) & 0xFFFFu) | (1u << 16));          // td1[31:16] | tile_dim0=1
  g1[4] = (int)(32u | (16u << 16));                            // tile_dim1=32 | tile_dim2=16
  g1[5] = (int)(unsigned)(stride0_elems & 0xFFFFFFFFull);      // dim0_stride[31:0]
  g1[6] = (int)(((stride0_elems >> 32) & 0xFFFFull)
              | ((stride1_elems & 0xFFFFull) << 16));          // d0s[47:32] | d1s[15:0]
  g1[7] = (int)((stride1_elems >> 16) & 0xFFFFFFFFull);        // d1s[47:16]

  i32x4 g2 = { (int)TD, 0, 0, 0 };    // tensor_dim2 huge; dim3/stride2/tile3 = 0
  i32x4 g3 = { 0, 0, 0, 0 };

  // ensure earlier DS reads of the destination buffer retired before DMA reuse
  asm volatile("s_wait_dscnt 0x0" ::: "memory");

#if __clang_major__ >= 23
  i32x8 gz = { 0, 0, 0, 0, 0, 0, 0, 0 };
  __builtin_amdgcn_tensor_load_to_lds(g0, g1, g2, g3, gz, 0);
#else
  __builtin_amdgcn_tensor_load_to_lds(g0, g1, g2, g3, 0);
#endif
}
#endif // HAVE_TDM

// ---------------------------------------------------------------------------
// mv_linear as 8 per-blade GEMMs: Y[n,o,i] = sum_m X[n,m,i] * W[o,m,grade(i)]
// One wave per 16x16 output tile per blade. TDM double-buffers the K tiles
// into LDS; fragments convert f32 -> bf16; v_wmma_f32_16x16x32_bf16 accum.
// ---------------------------------------------------------------------------
template <int MIN, int OUTD>
__global__ void __launch_bounds__(32)
mv_linear_wmma(const float* __restrict__ X, const float* __restrict__ W,
               const float* __restrict__ bias, const float* __restrict__ RES,
               float* __restrict__ Y)
{
  constexpr int NT = MIN / 32;           // K tiles
  __shared__ float Asf[2][16 * 33];      // [m][k], row stride 33 (TDM pad)
  __shared__ float Bsf[2][16 * 33];      // [o][k], row stride 33

  const int lane  = threadIdx.x;
  const int n0    = blockIdx.x * 16;
  const int o0    = blockIdx.y * 16;
  const int blade = blockIdx.z;
  const int GR[8] = {0, 1, 1, 1, 2, 2, 2, 3};
  const int g     = GR[blade];
  const int hf = lane >> 4;              // 0 or 1
  const int lr = lane & 15;

  auto issue = [&](int t, int buf) {
    const int k0 = t * 32;
#ifdef HAVE_TDM
    tdm_load_tile(&X[(((size_t)n0) * MIN + k0) * 8 + blade], &Asf[buf][0],
                  8ull, (unsigned long long)MIN * 8ull);
    tdm_load_tile(&W[(((size_t)o0) * MIN + k0) * 4 + g], &Bsf[buf][0],
                  4ull, (unsigned long long)MIN * 4ull);
#else
    #pragma unroll
    for (int e = 0; e < 16; ++e)
      Asf[buf][e * 33 + lane] = X[(((size_t)(n0 + e)) * MIN + (k0 + lane)) * 8 + blade];
    #pragma unroll
    for (int e = 0; e < 16; ++e) {
      int kk = e * 2 + hf;
      Bsf[buf][lr * 33 + kk] = W[(((size_t)(o0 + lr)) * MIN + (k0 + kk)) * 4 + g];
    }
#endif
  };

  v8f c = {0.f, 0.f, 0.f, 0.f, 0.f, 0.f, 0.f, 0.f};
  issue(0, 0);

  #pragma unroll
  for (int t = 0; t < NT; ++t) {
    const int cur = t & 1;
    if (t + 1 < NT) issue(t + 1, cur ^ 1);        // prefetch next tile via TDM
#ifdef HAVE_TDM
    if (t + 1 < NT) __builtin_amdgcn_s_wait_tensorcnt(2);   // current tile done
    else            __builtin_amdgcn_s_wait_tensorcnt(0);   // drain
#endif
    __syncthreads();

    // Fragments per ISA 7.12.2 (wave32, 16-bit operands):
    //   A: lane<16 -> M=lr, K in {0..7,16..23}; lane>=16 -> K in {8..15,24..31}
    //   B: lane<16 -> K in {0..7,16..23}; lane>=16 -> K in {8..15,24..31}; col N = lr
    union { v16bf v; unsigned short u[16]; } fa, fb;
    #pragma unroll
    for (int j = 0; j < 16; ++j) {
      int ak = (j < 8) ? (hf * 8 + j) : (8 + hf * 8 + j);   // == 16+hf*8+(j-8)
      int bk = (j < 8) ? (hf * 8 + j) : (8 + hf * 8 + j);
      fa.u[j] = f2bf(Asf[cur][lr * 33 + ak]);
      fb.u[j] = f2bf(Bsf[cur][lr * 33 + bk]);
    }
    c = __builtin_amdgcn_wmma_f32_16x16x32_bf16(false, fa.v, false, fb.v,
                                                (short)0, c, false, false);
    __syncthreads();
  }

  // C/D layout: element v -> row = v + 8*hf, col = lr
  #pragma unroll
  for (int vv = 0; vv < 8; ++vv) {
    int row = n0 + vv + hf * 8;
    int col = o0 + lr;
    size_t idx = (((size_t)row) * OUTD + col) * 8 + blade;
    float val = c[vv];
    if (bias && blade == 0) val += bias[col];
    if (RES) val += RES[idx];
    Y[idx] = val;
  }
}

// ---------------------------------------------------------------------------
// Multivector layernorm: y[n,c,:] = a[c] * x[n,c,:] / (mean_c ||x[n,c,:]|| + eps)
// optional post-residual: out = postres + LN(in)
// ---------------------------------------------------------------------------
__global__ void __launch_bounds__(128)
mv_ln_kernel(float* __restrict__ out, const float* __restrict__ in,
             const float* __restrict__ postres, const float* __restrict__ a)
{
  __shared__ float red[128];
  const int n = blockIdx.x, c = threadIdx.x;
  const size_t base = ((size_t)n * DCH + c) * 8;
  float x[8], ss = 0.f;
  #pragma unroll
  for (int i = 0; i < 8; ++i) { x[i] = in[base + i]; ss += x[i] * x[i]; }
  red[c] = sqrtf(ss);
  __syncthreads();
  for (int s = 64; s > 0; s >>= 1) {
    if (c < s) red[c] += red[c + s];
    __syncthreads();
  }
  const float mean = red[0] * (1.0f / 128.0f) + 1e-6f;
  const float sc = a[c] / mean;
  #pragma unroll
  for (int i = 0; i < 8; ++i) {
    float v = sc * x[i];
    if (postres) v += postres[base + i];
    out[base + i] = v;
  }
}

// ---------------------------------------------------------------------------
// Attention: per (batch, head); SEQ=25, features = dh*8 = 128 contiguous floats
// ---------------------------------------------------------------------------
__global__ void __launch_bounds__(256)
attention_kernel(const float* __restrict__ Q, const float* __restrict__ K,
                 const float* __restrict__ V, float* __restrict__ O)
{
  __shared__ float qs[SEQ][128], ks[SEQ][128], vs[SEQ][128], ps[SEQ][SEQ];
  const int b = blockIdx.x / HEADS, h = blockIdx.x % HEADS;
  const int tid = threadIdx.x;
  const size_t base = ((size_t)b * SEQ) * 1024 + (size_t)h * 128;

  for (int idx = tid; idx < SEQ * 128; idx += 256) {
    int s = idx >> 7, f = idx & 127;
    size_t gi = base + (size_t)s * 1024 + f;
    qs[s][f] = Q[gi]; ks[s][f] = K[gi]; vs[s][f] = V[gi];
  }
  __syncthreads();

  for (int idx = tid; idx < SEQ * SEQ; idx += 256) {
    int s = idx / SEQ, t = idx % SEQ;
    float acc = 0.f;
    #pragma unroll 8
    for (int f = 0; f < 128; ++f) acc += qs[s][f] * ks[t][f];
    ps[s][t] = acc * 0.0883883476483184f;   // 1/sqrt(128)
  }
  __syncthreads();

  if (tid < SEQ) {
    float mx = -1e30f;
    for (int t = 0; t < SEQ; ++t) mx = fmaxf(mx, ps[tid][t]);
    float sm = 0.f;
    for (int t = 0; t < SEQ; ++t) { float e = __expf(ps[tid][t] - mx); ps[tid][t] = e; sm += e; }
    float inv = 1.0f / sm;
    for (int t = 0; t < SEQ; ++t) ps[tid][t] *= inv;
  }
  __syncthreads();

  for (int idx = tid; idx < SEQ * 128; idx += 256) {
    int s = idx >> 7, f = idx & 127;
    float acc = 0.f;
    #pragma unroll
    for (int t = 0; t < SEQ; ++t) acc += ps[s][t] * vs[t][f];
    O[base + (size_t)s * 1024 + f] = acc;
  }
}

// ---------------------------------------------------------------------------
// Geometric product, Cl(3,0), blade order [1,e1,e2,e3,e12,e13,e23,e123]
// ---------------------------------------------------------------------------
__global__ void __launch_bounds__(256)
gp_kernel(const float* __restrict__ XL, const float* __restrict__ XR,
          float* __restrict__ G, long total)
{
  static const int PI[64] = {
    0,1,2,3,4,5,6,7,  1,0,4,5,2,3,7,6,  2,4,0,6,1,7,3,5,  3,5,6,0,7,1,2,4,
    4,2,1,7,0,6,5,3,  5,3,7,1,6,0,4,2,  6,7,3,2,5,4,0,1,  7,6,5,4,3,2,1,0 };
  static const float PS[64] = {
    1, 1, 1, 1, 1, 1, 1, 1,   1, 1, 1, 1, 1, 1, 1, 1,
    1,-1, 1, 1,-1,-1, 1,-1,   1,-1,-1, 1, 1,-1,-1, 1,
    1,-1, 1, 1,-1,-1, 1,-1,   1,-1,-1, 1, 1,-1,-1, 1,
    1, 1,-1, 1,-1, 1,-1,-1,   1, 1,-1, 1,-1, 1,-1,-1 };

  long idx = (long)blockIdx.x * 256 + threadIdx.x;
  if (idx >= total) return;
  const size_t base = (size_t)idx * 8;
  float a[8], b[8], g[8] = {0, 0, 0, 0, 0, 0, 0, 0};
  #pragma unroll
  for (int i = 0; i < 8; ++i) { a[i] = XL[base + i]; b[i] = XR[base + i]; }
  #pragma unroll
  for (int i = 0; i < 8; ++i)
    #pragma unroll
    for (int k = 0; k < 8; ++k)
      g[PI[i * 8 + k]] += PS[i * 8 + k] * a[i] * b[k];
  #pragma unroll
  for (int j = 0; j < 8; ++j) G[base + j] = g[j];
}

// ---------------------------------------------------------------------------
// MVSiLU: gate each grade by sigmoid(a*invariant + b)
// ---------------------------------------------------------------------------
__global__ void __launch_bounds__(256)
mvsilu_kernel(const float* __restrict__ X, const float* __restrict__ SA,
              const float* __restrict__ SB, float* __restrict__ Y,
              long total, int C)
{
  long idx = (long)blockIdx.x * 256 + threadIdx.x;
  if (idx >= total) return;
  const int c = (int)(idx % C);
  const size_t base = (size_t)idx * 8;
  float x[8];
  #pragma unroll
  for (int i = 0; i < 8; ++i) x[i] = X[base + i];
  float inv[4];
  inv[0] = x[0];
  inv[1] = x[1] * x[1] + x[2] * x[2] + x[3] * x[3];
  inv[2] = x[4] * x[4] + x[5] * x[5] + x[6] * x[6];
  inv[3] = x[7] * x[7];
  float s[4];
  #pragma unroll
  for (int gq = 0; gq < 4; ++gq) {
    float z = SA[c * 4 + gq] * inv[gq] + SB[c * 4 + gq];
    s[gq] = 1.0f / (1.0f + __expf(-z));
  }
  Y[base + 0] = s[0] * x[0];
  Y[base + 1] = s[1] * x[1]; Y[base + 2] = s[1] * x[2]; Y[base + 3] = s[1] * x[3];
  Y[base + 4] = s[2] * x[4]; Y[base + 5] = s[2] * x[5]; Y[base + 6] = s[2] * x[6];
  Y[base + 7] = s[3] * x[7];
}

// ---------------------------------------------------------------------------
extern "C" void kernel_launch(void* const* d_in, const int* in_sizes, int n_in,
                              void* d_out, int out_size, void* d_ws, size_t ws_size,
                              hipStream_t stream)
{
  const float* src  = (const float*)d_in[0];
  const float* ln1a = (const float*)d_in[1];
  const float* ln2a = (const float*)d_in[2];
  const float* ln3a = (const float*)d_in[3];
  const float* wq = (const float*)d_in[4];  const float* bq = (const float*)d_in[5];
  const float* wk = (const float*)d_in[6];  const float* bk = (const float*)d_in[7];
  const float* wv = (const float*)d_in[8];  const float* bv = (const float*)d_in[9];
  const float* wo = (const float*)d_in[10]; const float* bo = (const float*)d_in[11];
  const float* gw1 = (const float*)d_in[12]; const float* gb1 = (const float*)d_in[13];
  const float* gw2 = (const float*)d_in[14]; const float* gb2 = (const float*)d_in[15];
  const float* gw3 = (const float*)d_in[16]; const float* gb3 = (const float*)d_in[17];
  const float* gna = (const float*)d_in[18];
  const float* mw1 = (const float*)d_in[19]; const float* mb1 = (const float*)d_in[20];
  const float* sa  = (const float*)d_in[21]; const float* sb  = (const float*)d_in[22];
  const float* mw2 = (const float*)d_in[23]; const float* mb2 = (const float*)d_in[24];

  const int N  = in_sizes[0] / (DCH * 8);   // 25600 tokens
  const int Bb = N / SEQ;

  float* T  = (float*)d_ws;                 // [N,128,8]
  float* A1 = T  + (size_t)N * 1024;        // [N,128,8]
  float* A2 = A1 + (size_t)N * 1024;        // [N,128,8]
  float* P0 = A2 + (size_t)N * 1024;        // [N,256,8]
  float* P1 = P0 + (size_t)N * 2048;        // [N,256,8]
  float* Qb = P0;                           // alias: q,k in P0
  float* Kb = P0 + (size_t)N * 1024;
  float* Vb = P1;                           // alias: v,o in P1
  float* Ob = P1 + (size_t)N * 1024;
  float* out = (float*)d_out;

  dim3 g128(N / 16, 8, 8);    // OUT=128: (n tiles, o tiles, blades)
  dim3 g256(N / 16, 16, 8);   // OUT=256

  // x = src + attention(LN(src, ln1_a))
  mv_ln_kernel<<<N, 128, 0, stream>>>(T, src, nullptr, ln1a);
  mv_linear_wmma<128, 128><<<g128, 32, 0, stream>>>(T, wq, bq, nullptr, Qb);
  mv_linear_wmma<128, 128><<<g128, 32, 0, stream>>>(T, wk, bk, nullptr, Kb);
  mv_linear_wmma<128, 128><<<g128, 32, 0, stream>>>(T, wv, bv, nullptr, Vb);
  attention_kernel<<<Bb * HEADS, 256, 0, stream>>>(Qb, Kb, Vb, Ob);
  mv_linear_wmma<128, 128><<<g128, 32, 0, stream>>>(Ob, wo, bo, src, A1);

  // x = LN(x, ln2_a); GP layer; x = x + LN(gw3(gp(gw1 x, gw2 x)), gna)
  mv_ln_kernel<<<N, 128, 0, stream>>>(A2, A1, nullptr, ln2a);
  mv_linear_wmma<128, 256><<<g256, 32, 0, stream>>>(A2, gw1, gb1, nullptr, P0);
  mv_linear_wmma<128, 256><<<g256, 32, 0, stream>>>(A2, gw2, gb2, nullptr, P1);
  long tot = (long)N * 256;
  gp_kernel<<<(unsigned)((tot + 255) / 256), 256, 0, stream>>>(P0, P1, P0, tot);
  mv_linear_wmma<256, 128><<<g128, 32, 0, stream>>>(P0, gw3, gb3, nullptr, T);
  mv_ln_kernel<<<N, 128, 0, stream>>>(A1, T, A2, gna);       // A1 = A2 + LN(T,gna)

  // x = LN(x, ln3_a); out = x + mw2(silu(mw1 x))
  mv_ln_kernel<<<N, 128, 0, stream>>>(A2, A1, nullptr, ln3a);
  mv_linear_wmma<128, 256><<<g256, 32, 0, stream>>>(A2, mw1, mb1, nullptr, P0);
  mvsilu_kernel<<<(unsigned)((tot + 255) / 256), 256, 0, stream>>>(P0, sa, sb, P0, tot, 256);
  mv_linear_wmma<256, 128><<<g128, 32, 0, stream>>>(P0, mw2, mb2, A2, out);

  (void)n_in; (void)out_size; (void)ws_size;
}
